// SpectralMHSA_5600637354753
// MI455X (gfx1250) — matrix operand order repaired and checked by reading, not compiled, not run on hardware
//
#include <hip/hip_runtime.h>
#include <hip/hip_bf16.h>

// MHSA: N=4096 tokens, D=1024, H=16 heads, head_dim=64.
// f16 WMMA (16x16x32) with f32 accumulation; flash-attention with base-2 softmax.

typedef __attribute__((ext_vector_type(16))) _Float16 v16h;
typedef __attribute__((ext_vector_type(8)))  _Float16 v8h;
typedef __attribute__((ext_vector_type(4)))  _Float16 v4h;
typedef __attribute__((ext_vector_type(8)))  float    v8f;

#define NTOK   4096
#define DMODEL 1024
#define NHEAD  16
#define HDIM   64

// ---- fragment loaders -------------------------------------------------------
// A-matrix 16x32 f16: lane = M%16, kg = lane>>4; elements 0..15 -> K = kg*16 + e.
// p points at (row base) + kg*16 ; contiguous 16 halves.
__device__ __forceinline__ v16h fragA(const _Float16* p) {
  v8h lo = *(const v8h*)p;
  v8h hi = *(const v8h*)(p + 8);
  return __builtin_shufflevector(lo, hi, 0,1,2,3,4,5,6,7,8,9,10,11,12,13,14,15);
}
// B-matrix 32x16 f16: lane = N%16, kg = lane>>4; elements 0..15 -> K = kg*16 + e.
// p points at (column-row base) + kg*16 ; contiguous 16 halves.
__device__ __forceinline__ v16h fragB(const _Float16* p) {
  v8h lo = *(const v8h*)p;
  v8h hi = *(const v8h*)(p + 8);
  return __builtin_shufflevector(lo, hi, 0,1,2,3,4,5,6,7,8,9,10,11,12,13,14,15);
}

// ---- fp32 -> f16 conversion -------------------------------------------------
__global__ void cvt_f32_f16(const float* __restrict__ src, _Float16* __restrict__ dst, int n) {
  int i = (blockIdx.x * blockDim.x + threadIdx.x) * 4;
  if (i < n) {
    float4 f = *(const float4*)(src + i);
    v4h h;
    h[0] = (_Float16)f.x; h[1] = (_Float16)f.y;
    h[2] = (_Float16)f.z; h[3] = (_Float16)f.w;
    *(v4h*)(dst + i) = h;
  }
}

// ---- tiled GEMM: Y = A[M,1024] @ W[Nout,1024]^T + bias ----------------------
// Block tile 128(M) x 128(N), K-block 64. 8 waves as 4(M) x 2(N); each wave
// computes a 32x64 tile = 2x4 WMMA 16x16 tiles (16 WMMA per staged K-block).
// MODE 0: f16 row-major out [M, DMODEL]
// MODE 1: f16 transposed out Yt[n*NTOK + m]   (per-feature-transposed V)
// MODE 2: f32 row-major out (final projection -> d_out)
template <int MODE>
__global__ void __launch_bounds__(256)
gemm_xWT(const _Float16* __restrict__ A, const _Float16* __restrict__ W,
         const float* __restrict__ bias, void* __restrict__ Yv) {
  constexpr int KB  = 64;
  constexpr int LDA = KB + 8;  // 72 halves: 144B row stride, 16B aligned
  __shared__ _Float16 As[128 * LDA];
  __shared__ _Float16 Bs[128 * LDA];

  const int tid  = threadIdx.x;
  const int lane = tid & 31;
  const int wave = tid >> 5;
  const int wm   = wave & 3;   // M sub-tile (32 rows)
  const int wn   = wave >> 2;  // N sub-tile (64 cols)
  const int m0   = blockIdx.y * 128;
  const int n0   = blockIdx.x * 128;
  const int r16  = lane & 15;
  const int kg   = lane >> 4;

  v8f acc[2][4] = {};

  for (int k0 = 0; k0 < DMODEL; k0 += KB) {
    // stage A tile 128x64: 2 threads per row, 32 halves each
    {
      int r = tid & 127, c = (tid >> 7) * 32;
      const _Float16* g = A + (m0 + r) * DMODEL + k0 + c;
      _Float16* s = As + r * LDA + c;
      *(v8h*)(s +  0) = *(const v8h*)(g +  0);
      *(v8h*)(s +  8) = *(const v8h*)(g +  8);
      *(v8h*)(s + 16) = *(const v8h*)(g + 16);
      *(v8h*)(s + 24) = *(const v8h*)(g + 24);
    }
    // stage B(W) tile 128x64: 2 threads per row, 32 halves each
    {
      int r = tid & 127, c = (tid >> 7) * 32;
      const _Float16* g = W + (n0 + r) * DMODEL + k0 + c;
      _Float16* s = Bs + r * LDA + c;
      *(v8h*)(s +  0) = *(const v8h*)(g +  0);
      *(v8h*)(s +  8) = *(const v8h*)(g +  8);
      *(v8h*)(s + 16) = *(const v8h*)(g + 16);
      *(v8h*)(s + 24) = *(const v8h*)(g + 24);
    }
    __syncthreads();

#pragma unroll
    for (int kk = 0; kk < KB; kk += 32) {
      v16h af[2], bf[4];
#pragma unroll
      for (int i = 0; i < 2; ++i)
        af[i] = fragA(As + (wm * 32 + i * 16 + r16) * LDA + kk + kg * 16);
#pragma unroll
      for (int j = 0; j < 4; ++j)
        bf[j] = fragB(Bs + (wn * 64 + j * 16 + r16) * LDA + kk + kg * 16);
#pragma unroll
      for (int i = 0; i < 2; ++i)
#pragma unroll
        for (int j = 0; j < 4; ++j)
          acc[i][j] = __builtin_amdgcn_wmma_f32_16x16x32_f16(
              false, af[i], false, bf[j], (short)0, acc[i][j], false, false);
    }
    __syncthreads();
  }

  // C layout: VGPR r, lane L -> M = r + 8*(L>>4), N = L&15
#pragma unroll
  for (int i = 0; i < 2; ++i) {
#pragma unroll
    for (int j = 0; j < 4; ++j) {
      int mbase = m0 + wm * 32 + i * 16 + 8 * kg;
      int nidx  = n0 + wn * 64 + j * 16 + r16;
      float b = bias[nidx];
#pragma unroll
      for (int r = 0; r < 8; ++r) {
        float v = acc[i][j][r] + b;
        int m = mbase + r;
        if (MODE == 0)      ((_Float16*)Yv)[m * DMODEL + nidx] = (_Float16)v;
        else if (MODE == 1) ((_Float16*)Yv)[nidx * NTOK + m]   = (_Float16)v;
        else                ((float*)Yv)[m * DMODEL + nidx]    = v;
      }
    }
  }
}

// ---- flash attention --------------------------------------------------------
// Grid: (NTOK/128, NHEAD). 8 waves/block; each wave owns 16 query rows.
// Key step = 64 keys: S = QK^T (8 WMMA), base-2 online softmax, ctx += P V (8 WMMA).
__global__ void __launch_bounds__(256)
attn_flash(const _Float16* __restrict__ Q, const _Float16* __restrict__ K,
           const _Float16* __restrict__ Vt, _Float16* __restrict__ ctx) {
  __shared__ _Float16 Pls[8 * 16 * 64];  // wave-private P transpose tiles (2KB/wave)

  const int tid  = threadIdx.x;
  const int lane = tid & 31;
  const int wave = tid >> 5;
  const int h    = blockIdx.y;
  const int q0   = blockIdx.x * 128 + wave * 16;
  const int r16  = lane & 15;
  const int kg   = lane >> 4;
  const int hoff = h * HDIM;
  _Float16* Pw = Pls + wave * (16 * 64);

  // Q A-fragments for this wave's 16 rows (head dim 64 = two 16x32 frags)
  v16h qf[2];
#pragma unroll
  for (int t = 0; t < 2; ++t)
    qf[t] = fragA(Q + (q0 + r16) * DMODEL + hoff + t * 32 + kg * 16);

  float mrow[8], lrow[8];
#pragma unroll
  for (int r = 0; r < 8; ++r) { mrow[r] = -3.0e38f; lrow[r] = 0.0f; }
  v8f cacc[4] = {};  // 16 rows x 64 cols of context, 4 col-tiles

  // base-2 softmax: fold 1/sqrt(64) * log2(e) into the score scaling
  const float sl2e = 0.125f * 1.4426950408889634f;

  for (int j0 = 0; j0 < NTOK; j0 += 64) {
    // prefetch next key/value tiles into L2/L0 (global_prefetch_b8)
    if (j0 + 64 < NTOK) {
      __builtin_prefetch(K + (j0 + 64 + r16 * 4) * DMODEL + hoff, 0, 1);
      __builtin_prefetch(Vt + (hoff + r16 * 4) * NTOK + j0 + 64, 0, 1);
    }

    // S = Q K^T : 4 key-column tiles x chained K-dim (64) WMMAs
    v8f s[4] = {};
#pragma unroll
    for (int ct = 0; ct < 4; ++ct)
#pragma unroll
      for (int t = 0; t < 2; ++t) {
        v16h bf = fragB(K + (j0 + ct * 16 + r16) * DMODEL + hoff + t * 32 + kg * 16);
        s[ct] = __builtin_amdgcn_wmma_f32_16x16x32_f16(
            false, qf[t], false, bf, (short)0, s[ct], false, false);
      }

    // online softmax in log2 domain; row = r + 8*kg
    float alpha[8];
#pragma unroll
    for (int r = 0; r < 8; ++r) {
      float s0 = s[0][r] * sl2e;
      float s1 = s[1][r] * sl2e;
      float s2 = s[2][r] * sl2e;
      float s3 = s[3][r] * sl2e;
      float tmax = fmaxf(fmaxf(s0, s1), fmaxf(s2, s3));
#pragma unroll
      for (int xm = 8; xm >= 1; xm >>= 1)
        tmax = fmaxf(tmax, __shfl_xor(tmax, xm, 32));
      float mnew = fmaxf(mrow[r], tmax);
      float a  = exp2f(mrow[r] - mnew);
      float p0 = exp2f(s0 - mnew);
      float p1 = exp2f(s1 - mnew);
      float p2 = exp2f(s2 - mnew);
      float p3 = exp2f(s3 - mnew);
      float rs = (p0 + p1) + (p2 + p3);
#pragma unroll
      for (int xm = 8; xm >= 1; xm >>= 1)
        rs += __shfl_xor(rs, xm, 32);
      lrow[r] = lrow[r] * a + rs;
      mrow[r] = mnew;
      alpha[r] = a;
      s[0][r] = p0; s[1][r] = p1; s[2][r] = p2; s[3][r] = p3;
    }

    // P: C-layout -> row-major 16x64 in LDS (wave-private, DS ops in order)
#pragma unroll
    for (int ct = 0; ct < 4; ++ct)
#pragma unroll
      for (int r = 0; r < 8; ++r)
        Pw[(r + 8 * kg) * 64 + ct * 16 + r16] = (_Float16)s[ct][r];

    // read back as A-fragments (16 rows x 64 keys = two 16x32 frags)
    v16h pf[2];
#pragma unroll
    for (int t = 0; t < 2; ++t)
      pf[t] = fragA(Pw + r16 * 64 + t * 32 + kg * 16);

    // rescale context accumulators and accumulate P @ V
#pragma unroll
    for (int c = 0; c < 4; ++c)
#pragma unroll
      for (int r = 0; r < 8; ++r)
        cacc[c][r] *= alpha[r];
#pragma unroll
    for (int c = 0; c < 4; ++c)
#pragma unroll
      for (int t = 0; t < 2; ++t) {
        v16h vf = fragB(Vt + (hoff + c * 16 + r16) * NTOK + j0 + t * 32 + kg * 16);
        cacc[c] = __builtin_amdgcn_wmma_f32_16x16x32_f16(
            false, pf[t], false, vf, (short)0, cacc[c], false, false);
      }
  }

  // normalize and store f16 context [NTOK][DMODEL]
#pragma unroll
  for (int r = 0; r < 8; ++r) lrow[r] = 1.0f / lrow[r];
#pragma unroll
  for (int c = 0; c < 4; ++c)
#pragma unroll
    for (int r = 0; r < 8; ++r) {
      float v = cacc[c][r] * lrow[r];
      int m = q0 + r + 8 * kg;
      int n = hoff + c * 16 + r16;
      ctx[m * DMODEL + n] = (_Float16)v;
    }
}

// ---- launch -----------------------------------------------------------------
extern "C" void kernel_launch(void* const* d_in, const int* in_sizes, int n_in,
                              void* d_out, int out_size, void* d_ws, size_t ws_size,
                              hipStream_t stream) {
  (void)in_sizes; (void)n_in; (void)out_size; (void)ws_size;
  const float* x  = (const float*)d_in[0];
  const float* Wq = (const float*)d_in[1];
  const float* bq = (const float*)d_in[2];
  const float* Wk = (const float*)d_in[3];
  const float* bk = (const float*)d_in[4];
  const float* Wv = (const float*)d_in[5];
  const float* bv = (const float*)d_in[6];
  const float* Wo = (const float*)d_in[7];
  const float* bo = (const float*)d_in[8];

  // workspace layout (f16), 48 MiB total
  char* ws = (char*)d_ws;
  _Float16* xh   = (_Float16*)(ws + (size_t)0);
  _Float16* Wqh  = (_Float16*)(ws + ((size_t)8  << 20));
  _Float16* Wkh  = (_Float16*)(ws + ((size_t)10 << 20));
  _Float16* Wvh  = (_Float16*)(ws + ((size_t)12 << 20));
  _Float16* Woh  = (_Float16*)(ws + ((size_t)14 << 20));
  _Float16* Qh   = (_Float16*)(ws + ((size_t)16 << 20));
  _Float16* Kh   = (_Float16*)(ws + ((size_t)24 << 20));
  _Float16* Vt   = (_Float16*)(ws + ((size_t)32 << 20));  // [DMODEL][NTOK]
  _Float16* ctxh = (_Float16*)(ws + ((size_t)40 << 20));

  const int XE = NTOK * DMODEL;    // 4 Mi elems
  const int WE = DMODEL * DMODEL;  // 1 Mi elems
  cvt_f32_f16<<<(XE / 4 + 255) / 256, 256, 0, stream>>>(x,  xh,  XE);
  cvt_f32_f16<<<(WE / 4 + 255) / 256, 256, 0, stream>>>(Wq, Wqh, WE);
  cvt_f32_f16<<<(WE / 4 + 255) / 256, 256, 0, stream>>>(Wk, Wkh, WE);
  cvt_f32_f16<<<(WE / 4 + 255) / 256, 256, 0, stream>>>(Wv, Wvh, WE);
  cvt_f32_f16<<<(WE / 4 + 255) / 256, 256, 0, stream>>>(Wo, Woh, WE);

  dim3 ggrid(DMODEL / 128, NTOK / 128);
  gemm_xWT<0><<<ggrid, 256, 0, stream>>>(xh, Wqh, bq, (void*)Qh);
  gemm_xWT<0><<<ggrid, 256, 0, stream>>>(xh, Wkh, bk, (void*)Kh);
  gemm_xWT<1><<<ggrid, 256, 0, stream>>>(xh, Wvh, bv, (void*)Vt);

  attn_flash<<<dim3(NTOK / 128, NHEAD), 256, 0, stream>>>(Qh, Kh, Vt, ctxh);

  gemm_xWT<2><<<ggrid, 256, 0, stream>>>(ctxh, Woh, bo, d_out);
}
